// GNN_85074712199863
// MI455X (gfx1250) — matrix operand order repaired and checked by reading, not compiled, not run on hardware
//
#include <hip/hip_runtime.h>
#include <hip/hip_bf16.h>
#include <math.h>

// ---------------------------------------------------------------------------
// GNN (2x TransformerConv + link MLP) for gfx1250 (MI455X, wave32, WMMA).
// Dense GEMMs: V_WMMA_F32_16X16X32_BF16, fp32 accumulate.
// A-tiles: two contiguous float4-pairs per lane per K-step (b128 loads),
//          converted once and reused across all 8 output column tiles.
// B-tiles: weights pre-packed to bf16 in WMMA B layout (32B/lane/step).
// ---------------------------------------------------------------------------

typedef __attribute__((ext_vector_type(16))) __bf16 v16bf;
typedef __attribute__((ext_vector_type(8)))  float  v8f;

#define HID 128   // hidden width / GEMM output columns (all GEMMs)
#define NT  8     // HID/16 column tiles
#define GQ  128   // number of graphs

// ---- ordered-uint float max (for segment max via atomicMax) ----------------
__device__ __forceinline__ unsigned f2ord(float f) {
    unsigned u = __float_as_uint(f);
    return (u & 0x80000000u) ? ~u : (u | 0x80000000u);
}
__device__ __forceinline__ float ord2f(unsigned u) {
    return (u & 0x80000000u) ? __uint_as_float(u & 0x7fffffffu)
                             : __uint_as_float(~u);
}

// Load 8 contiguous floats (two b128) and convert into a[base..base+7]
__device__ __forceinline__ void put8(v16bf& a, const int base,
                                     const float* __restrict__ p) {
    const float4 u = reinterpret_cast<const float4*>(const_cast<float*>(p))[0];
    const float4 v = reinterpret_cast<const float4*>(const_cast<float*>(p))[1];
    a[base + 0] = (__bf16)u.x; a[base + 1] = (__bf16)u.y;
    a[base + 2] = (__bf16)u.z; a[base + 3] = (__bf16)u.w;
    a[base + 4] = (__bf16)v.x; a[base + 5] = (__bf16)v.y;
    a[base + 6] = (__bf16)v.z; a[base + 7] = (__bf16)v.w;
}
__device__ __forceinline__ void zero8(v16bf& a, const int base) {
#pragma unroll
    for (int i = 0; i < 8; ++i) a[base + i] = (__bf16)0.0f;
}

// ---------------------------------------------------------------------------
// Pack W[K x HID] (f32, row-major) into bf16 WMMA-B layout:
//   pb[((ks*NT + nt)*32 + lane)*16 + i] = bf16( W[(ks*32 + 8*(lane/16) + i + 8*(i/8))*HID
//                                                  + nt*16 + (lane%16)] )
// Each lane then loads its whole B fragment as 32 contiguous bytes.
// ---------------------------------------------------------------------------
__global__ void pack_b_bf16(const float* __restrict__ W,
                            __bf16* __restrict__ pb, int K) {
    const int KS = (K + 31) / 32;
    const int total = KS * NT * 32 * 16;
    int idx = blockIdx.x * blockDim.x + threadIdx.x;
    if (idx >= total) return;
    int i    = idx & 15;
    int lane = (idx >> 4) & 31;
    int rest = idx >> 9;
    int nt   = rest % NT;
    int ks   = rest / NT;
    int k = ks * 32 + ((lane >> 4) & 1) * 8 + i + (i & 8);
    int n = nt * 16 + (lane & 15);
    pb[idx] = (__bf16)((k < K) ? W[(size_t)k * HID + n] : 0.0f);
}

// ---------------------------------------------------------------------------
// GEMM: out[M x HID] = A[M x K] @ W[K x HID] (+ bias).
// One wave computes a full 16 x 128 output strip: the converted A fragment
// is reused across all 8 column tiles (8x less A traffic + cvt work).
// K is compile-time => all guards fold; A loads are 4x b128 per K-step.
// ---------------------------------------------------------------------------
template <int K>
__global__ void gemm_bias_wmma(const float* __restrict__ A, int lda,
                               const __bf16* __restrict__ Bp,
                               const float* __restrict__ bias,
                               float* __restrict__ out) {
    const int lane  = threadIdx.x;           // 0..31, EXEC all ones
    const int row0  = blockIdx.x * 16;
    const int khalf = (lane >> 4) & 1;
    const int mrow  = row0 + (lane & 15);
    const float* Arow = A + (size_t)mrow * lda;

    constexpr int KS = (K + 31) / 32;
    v8f acc[NT] = {};
#pragma unroll
    for (int ks = 0; ks < KS; ++ks) {
        const int kb = ks * 32;
        v16bf a;
        if (kb < K)      put8(a, 0, Arow + kb + khalf * 8);      else zero8(a, 0);
        if (kb + 16 < K) put8(a, 8, Arow + kb + 16 + khalf * 8); else zero8(a, 8);
#pragma unroll
        for (int nt = 0; nt < NT; ++nt) {
            const v16bf b = *reinterpret_cast<const v16bf*>(
                Bp + ((size_t)(ks * NT + nt) * 32 + lane) * 16);
            acc[nt] = __builtin_amdgcn_wmma_f32_16x16x32_bf16(
                          false, a, false, b, (short)0, acc[nt], false, false);
        }
    }
#pragma unroll
    for (int nt = 0; nt < NT; ++nt) {
        const int ncol = nt * 16 + (lane & 15);
        const float bv = bias ? bias[ncol] : 0.0f;
#pragma unroll
        for (int r = 0; r < 8; ++r) {
            int row = row0 + r + 8 * khalf;
            out[(size_t)row * HID + ncol] = acc[nt][r] + bv;
        }
    }
}

// ---------------------------------------------------------------------------
// Attention pass 1: logits + segment max over destination nodes
// ---------------------------------------------------------------------------
__global__ void attn_logits_max(const float* __restrict__ q,
                                const float* __restrict__ k,
                                const float* __restrict__ ef,
                                const int* __restrict__ src,
                                const int* __restrict__ dst,
                                float* __restrict__ logits,
                                unsigned* __restrict__ mbuf,
                                int E_, int H, int dh, float scale) {
    int idx = blockIdx.x * blockDim.x + threadIdx.x;
    if (idx >= E_ * H) return;
    int e = idx / H, h = idx - e * H;
    int s = src[e], d = dst[e];
    const float* qp = q  + (size_t)d * HID + h * dh;
    const float* kp = k  + (size_t)s * HID + h * dh;
    const float* ep = ef + (size_t)e * HID + h * dh;
    float acc = 0.0f;
    for (int i = 0; i < dh; ++i) acc = fmaf(qp[i], kp[i] + ep[i], acc);
    float lg = acc * scale;
    logits[idx] = lg;
    atomicMax(mbuf + (size_t)d * H + h, f2ord(lg));
}

// Attention pass 2: exp(logit - max) + segment-sum denominator
__global__ void attn_expsum(float* __restrict__ logits,
                            const unsigned* __restrict__ mbuf,
                            float* __restrict__ den,
                            const int* __restrict__ dst,
                            int E_, int H) {
    int idx = blockIdx.x * blockDim.x + threadIdx.x;
    if (idx >= E_ * H) return;
    int e = idx / H, h = idx - e * H;
    int d = dst[e];
    float m  = ord2f(mbuf[(size_t)d * H + h]);
    float ex = __expf(logits[idx] - m);
    logits[idx] = ex;                      // reuse buffer: now holds exp term
    atomicAdd(den + (size_t)d * H + h, ex);
}

// Attention pass 3: att[dst] += alpha * (v[src] + e)
__global__ void attn_aggregate(const float* __restrict__ exv,
                               const float* __restrict__ den,
                               const float* __restrict__ v,
                               const float* __restrict__ ef,
                               const int* __restrict__ src,
                               const int* __restrict__ dst,
                               float* __restrict__ att,
                               int E_, int H, int dh) {
    int idx = blockIdx.x * blockDim.x + threadIdx.x;
    if (idx >= E_ * H) return;
    int e = idx / H, h = idx - e * H;
    int s = src[e], d = dst[e];
    float alpha = exv[idx] / den[(size_t)d * H + h];
    const float* vp = v  + (size_t)s * HID + h * dh;
    const float* ep = ef + (size_t)e * HID + h * dh;
    float* ap       = att + (size_t)d * HID + h * dh;
    for (int i = 0; i < dh; ++i) atomicAdd(ap + i, alpha * (vp[i] + ep[i]));
}

// h = act(att + skip)   (slope=0.01 -> leaky relu, slope=1.0 -> identity)
__global__ void add_skip_act(const float* __restrict__ att,
                             const float* __restrict__ skip,
                             float* __restrict__ out, int n, float slope) {
    int i = blockIdx.x * blockDim.x + threadIdx.x;
    if (i >= n) return;
    float v = att[i] + skip[i];
    out[i] = (v > 0.0f) ? v : slope * v;
}

// Global mean pool: one thread per (node, dim)
__global__ void pool_accum(const float* __restrict__ h,
                           const int* __restrict__ batch,
                           float* __restrict__ psum,
                           float* __restrict__ pcnt, int N_) {
    int idx = blockIdx.x * blockDim.x + threadIdx.x;
    if (idx >= N_ * HID) return;
    int n = idx >> 7, d = idx & 127;
    int g = batch[n];
    atomicAdd(psum + (size_t)g * HID + d, h[idx]);
    if (d == 0) atomicAdd(pcnt + g, 1.0f);
}

__global__ void pool_div(const float* __restrict__ psum,
                         const float* __restrict__ pcnt,
                         float* __restrict__ pool, int total) {
    int i = blockIdx.x * blockDim.x + threadIdx.x;
    if (i >= total) return;
    pool[i] = psum[i] / fmaxf(pcnt[i >> 7], 1.0f);
}

__global__ void zero_f32(float* __restrict__ p, size_t n) {
    size_t i = (size_t)blockIdx.x * blockDim.x + threadIdx.x;
    size_t stride = (size_t)gridDim.x * blockDim.x;
    for (; i < n; i += stride) p[i] = 0.0f;
}

// ---------------------------------------------------------------------------
// Fused link-classifier MLP:
//   out[e] = relu( feat[e,0:400] @ W1 + b1 ) @ W2 + b2
//   feat = [ h[src] (128) | h[dst] (128) | ea (16) | pool[batch[src]] (128) ]
// One wave = 16 edges; K-loop fully unrolled (13 steps) so the piecewise
// segment selection (boundaries 128/256/272/400, all 16-aligned => each
// 8-float run stays inside one segment) folds to constant base pointers.
// 8 hidden n-tiles accumulate in registers => 104 static WMMAs here.
// ---------------------------------------------------------------------------
__global__ void mlp_fused_wmma(const float* __restrict__ h,
                               const float* __restrict__ ea,
                               const float* __restrict__ pool,
                               const int* __restrict__ src,
                               const int* __restrict__ dst,
                               const int* __restrict__ batch,
                               const __bf16* __restrict__ W1p,
                               const float* __restrict__ b1,
                               const float* __restrict__ W2,
                               const float* __restrict__ b2,
                               float* __restrict__ out) {
    const int lane  = threadIdx.x;          // 0..31
    const int tile  = blockIdx.x;
    const int erow  = tile * 16 + (lane & 15);
    const int khalf = (lane >> 4) & 1;

    const int sp = src[erow], dp = dst[erow];
    const float* fs = h    + (size_t)sp * HID;            // k in [0,128)
    const float* fd = h    + (size_t)dp * HID;            // k in [128,256)
    const float* fe = ea   + (size_t)erow * 16;           // k in [256,272)
    const float* fg = pool + (size_t)batch[sp] * HID;     // k in [272,400)

    v8f acc[NT] = {};
#pragma unroll
    for (int ks = 0; ks < 13; ++ks) {
        const int kb = ks * 32;
        v16bf a;
#pragma unroll
        for (int half16 = 0; half16 < 2; ++half16) {
            const int t = kb + 16 * half16;               // compile-time
            const int base = half16 * 8;
            if (t >= 400) { zero8(a, base); continue; }
            const float* segp;
            if      (t < 128) segp = fs + t;
            else if (t < 256) segp = fd + (t - 128);
            else if (t < 272) segp = fe + (t - 256);
            else              segp = fg + (t - 272);
            put8(a, base, segp + khalf * 8);
        }
#pragma unroll
        for (int nt = 0; nt < NT; ++nt) {
            const v16bf b = *reinterpret_cast<const v16bf*>(
                W1p + ((size_t)(ks * NT + nt) * 32 + lane) * 16);
            acc[nt] = __builtin_amdgcn_wmma_f32_16x16x32_bf16(
                          false, a, false, b, (short)0, acc[nt], false, false);
        }
    }

    // epilogue: bias + relu + dot(W2) per lane, reduce 16 columns per half
    float sr[8];
#pragma unroll
    for (int r = 0; r < 8; ++r) sr[r] = 0.0f;
#pragma unroll
    for (int nt = 0; nt < NT; ++nt) {
        int n = nt * 16 + (lane & 15);
        float bias = b1[n], w2 = W2[n];
#pragma unroll
        for (int r = 0; r < 8; ++r) {
            float hv = acc[nt][r] + bias;
            hv = (hv > 0.0f) ? hv : 0.0f;
            sr[r] += hv * w2;
        }
    }
    const float b2v = b2[0];
#pragma unroll
    for (int r = 0; r < 8; ++r) {
        float s = sr[r];
        s += __shfl_xor(s, 1);   // masks < 16: stays within each 16-lane half,
        s += __shfl_xor(s, 2);   // matching the D-matrix column striping
        s += __shfl_xor(s, 4);
        s += __shfl_xor(s, 8);
        if ((lane & 15) == 0)
            out[tile * 16 + r + 8 * khalf] = s + b2v;
    }
}

// ---------------------------------------------------------------------------
extern "C" void kernel_launch(void* const* d_in, const int* in_sizes, int n_in,
                              void* d_out, int out_size, void* d_ws, size_t ws_size,
                              hipStream_t stream) {
    (void)n_in; (void)out_size; (void)ws_size;

    const float* x     = (const float*)d_in[0];
    const int*   ei    = (const int*)  d_in[1];
    const float* ea    = (const float*)d_in[2];
    const int*   batch = (const int*)  d_in[3];
    const float *Wq1=(const float*)d_in[4],  *bq1=(const float*)d_in[5];
    const float *Wk1=(const float*)d_in[6],  *bk1=(const float*)d_in[7];
    const float *Wv1=(const float*)d_in[8],  *bv1=(const float*)d_in[9];
    const float *We1=(const float*)d_in[10];
    const float *Ws1=(const float*)d_in[11], *bs1=(const float*)d_in[12];
    const float *Wq2=(const float*)d_in[13], *bq2=(const float*)d_in[14];
    const float *Wk2=(const float*)d_in[15], *bk2=(const float*)d_in[16];
    const float *Wv2=(const float*)d_in[17], *bv2=(const float*)d_in[18];
    const float *We2=(const float*)d_in[19];
    const float *Ws2=(const float*)d_in[20], *bs2=(const float*)d_in[21];
    const float *W1 =(const float*)d_in[22], *b1 =(const float*)d_in[23];
    const float *W2 =(const float*)d_in[24], *b2 =(const float*)d_in[25];

    const int N = in_sizes[0] / 64;    // 50000 (multiple of 16)
    const int E = in_sizes[1] / 2;     // 400000 (multiple of 16)
    const int* src = ei;
    const int* dst = ei + E;

    // ---- workspace partition (floats; keep 32B alignment for packs) ----
    float* ws = (float*)d_ws;
    size_t off = 0;
    auto take = [&](size_t n) { float* p = ws + off; off += (n + 15) & ~15ull; return p; };
    float* q1 = take((size_t)N * HID);
    float* k1 = take((size_t)N * HID);
    float* v1 = take((size_t)N * HID);
    float* s1 = take((size_t)N * HID);
    float* e1 = take((size_t)E * HID);
    float* lg1 = take((size_t)E * 8);
    float* q2 = take((size_t)N * HID);
    float* k2 = take((size_t)N * HID);
    float* v2 = take((size_t)N * HID);
    float* s2 = take((size_t)N * HID);
    float* e2 = take((size_t)E * HID);
    float* lg2 = take((size_t)E);
    float* h1 = take((size_t)N * HID);
    float* h2 = take((size_t)N * HID);
    float* poolavg = take((size_t)GQ * HID);
    // packed bf16 weight buffers (sizes in floats = bf16/2)
    auto KSf = [](int K) { return (size_t)((K + 31) / 32) * NT * 32 * 16 / 2; };
    __bf16* pq1 = (__bf16*)take(KSf(64));
    __bf16* pk1 = (__bf16*)take(KSf(64));
    __bf16* pv1 = (__bf16*)take(KSf(64));
    __bf16* ps1 = (__bf16*)take(KSf(64));
    __bf16* pe1 = (__bf16*)take(KSf(16));
    __bf16* pq2 = (__bf16*)take(KSf(128));
    __bf16* pk2 = (__bf16*)take(KSf(128));
    __bf16* pv2 = (__bf16*)take(KSf(128));
    __bf16* ps2 = (__bf16*)take(KSf(128));
    __bf16* pe2 = (__bf16*)take(KSf(16));
    __bf16* pW1 = (__bf16*)take(KSf(400));
    // zero-initialized block (contiguous)
    float* zbase = ws + off;
    float*    att1 = take((size_t)N * HID);
    unsigned* m1   = (unsigned*)take((size_t)N * 8);
    float*    den1 = take((size_t)N * 8);
    float*    att2 = take((size_t)N * HID);
    unsigned* m2   = (unsigned*)take((size_t)N);
    float*    den2 = take((size_t)N);
    float*    psum = take((size_t)GQ * HID);
    float*    pcnt = take((size_t)GQ);
    size_t zcount = (size_t)(ws + off - zbase);

    const dim3 wv(32);
    const dim3 gN(N / 16);
    const dim3 gE(E / 16);
    const int  TB = 256;

    // 0) clear accumulators / maxima / denominators / pool
    zero_f32<<<2048, TB, 0, stream>>>(zbase, zcount);

    // 0b) pack all weight matrices into bf16 WMMA-B layout
    auto packN = [](int K) { return ((K + 31) / 32) * NT * 32 * 16; };
    pack_b_bf16<<<(packN(64)  + TB - 1) / TB, TB, 0, stream>>>(Wq1, pq1, 64);
    pack_b_bf16<<<(packN(64)  + TB - 1) / TB, TB, 0, stream>>>(Wk1, pk1, 64);
    pack_b_bf16<<<(packN(64)  + TB - 1) / TB, TB, 0, stream>>>(Wv1, pv1, 64);
    pack_b_bf16<<<(packN(64)  + TB - 1) / TB, TB, 0, stream>>>(Ws1, ps1, 64);
    pack_b_bf16<<<(packN(16)  + TB - 1) / TB, TB, 0, stream>>>(We1, pe1, 16);
    pack_b_bf16<<<(packN(128) + TB - 1) / TB, TB, 0, stream>>>(Wq2, pq2, 128);
    pack_b_bf16<<<(packN(128) + TB - 1) / TB, TB, 0, stream>>>(Wk2, pk2, 128);
    pack_b_bf16<<<(packN(128) + TB - 1) / TB, TB, 0, stream>>>(Wv2, pv2, 128);
    pack_b_bf16<<<(packN(128) + TB - 1) / TB, TB, 0, stream>>>(Ws2, ps2, 128);
    pack_b_bf16<<<(packN(16)  + TB - 1) / TB, TB, 0, stream>>>(We2, pe2, 16);
    pack_b_bf16<<<(packN(400) + TB - 1) / TB, TB, 0, stream>>>(W1, pW1, 400);

    // 1) layer-1 projections (K=64) + edge features (K=16)
    gemm_bias_wmma<64><<<gN, wv, 0, stream>>>(x, 64, pq1, bq1, q1);
    gemm_bias_wmma<64><<<gN, wv, 0, stream>>>(x, 64, pk1, bk1, k1);
    gemm_bias_wmma<64><<<gN, wv, 0, stream>>>(x, 64, pv1, bv1, v1);
    gemm_bias_wmma<64><<<gN, wv, 0, stream>>>(x, 64, ps1, bs1, s1);
    gemm_bias_wmma<16><<<gE, wv, 0, stream>>>(ea, 16, pe1, nullptr, e1);

    // 2) layer-1 attention (H=8, dh=16, scale=1/sqrt(16))
    int T1 = E * 8;
    attn_logits_max<<<(T1 + TB - 1) / TB, TB, 0, stream>>>(
        q1, k1, e1, src, dst, lg1, m1, E, 8, 16, 0.25f);
    attn_expsum<<<(T1 + TB - 1) / TB, TB, 0, stream>>>(lg1, m1, den1, dst, E, 8);
    attn_aggregate<<<(T1 + TB - 1) / TB, TB, 0, stream>>>(
        lg1, den1, v1, e1, src, dst, att1, E, 8, 16);
    add_skip_act<<<((size_t)N * HID + TB - 1) / TB, TB, 0, stream>>>(
        att1, s1, h1, N * HID, 0.01f);               // leaky relu

    // 3) layer-2 projections (K=128) + edge features
    gemm_bias_wmma<128><<<gN, wv, 0, stream>>>(h1, HID, pq2, bq2, q2);
    gemm_bias_wmma<128><<<gN, wv, 0, stream>>>(h1, HID, pk2, bk2, k2);
    gemm_bias_wmma<128><<<gN, wv, 0, stream>>>(h1, HID, pv2, bv2, v2);
    gemm_bias_wmma<128><<<gN, wv, 0, stream>>>(h1, HID, ps2, bs2, s2);
    gemm_bias_wmma<16><<<gE, wv, 0, stream>>>(ea, 16, pe2, nullptr, e2);

    // 4) layer-2 attention (H=1, dh=128, scale=1/sqrt(128))
    const float sc2 = 0.088388347648318447f;
    attn_logits_max<<<(E + TB - 1) / TB, TB, 0, stream>>>(
        q2, k2, e2, src, dst, lg2, m2, E, 1, HID, sc2);
    attn_expsum<<<(E + TB - 1) / TB, TB, 0, stream>>>(lg2, m2, den2, dst, E, 1);
    attn_aggregate<<<(E + TB - 1) / TB, TB, 0, stream>>>(
        lg2, den2, v2, e2, src, dst, att2, E, 1, HID);
    add_skip_act<<<((size_t)N * HID + TB - 1) / TB, TB, 0, stream>>>(
        att2, s2, h2, N * HID, 1.0f);                // identity

    // 5) global mean pool
    pool_accum<<<((size_t)N * HID + TB - 1) / TB, TB, 0, stream>>>(
        h2, batch, psum, pcnt, N);
    pool_div<<<(GQ * HID + TB - 1) / TB, TB, 0, stream>>>(psum, pcnt, poolavg,
                                                          GQ * HID);

    // 6) fused link MLP (gather + [E,400]@[400,128] + relu + @[128,1])
    mlp_fused_wmma<<<E / 16, wv, 0, stream>>>(h2, ea, poolavg, src, dst, batch,
                                              pW1, b1, W2, b2, (float*)d_out);
}
